// FPN3DWithMultiHeadAttention_20392504721973
// MI455X (gfx1250) — compile-verified, matrix operand order checked by reading, not run on hardware
//
#include <hip/hip_runtime.h>
#include <hip/hip_bf16.h>

// ---------------------------------------------------------------------------
// MI455X (gfx1250): all matmuls via v_wmma_f32_16x16x32_bf16.
// Weights pre-packed to bf16 in GEMM tile layout, activations carried as
// bf16 copies for WMMA consumers (f32 kept only for residual adds), LDS
// tile staging via GLOBAL_LOAD_ASYNC_TO_LDS_B128 (ASYNCcnt) when available.
// ---------------------------------------------------------------------------

typedef __bf16 bf16;
typedef unsigned short u16;
typedef __attribute__((ext_vector_type(16))) __bf16 bf16x16;
typedef __attribute__((ext_vector_type(8)))  __bf16 bf16x8;
typedef __attribute__((ext_vector_type(8)))  float  f32x8;
typedef __attribute__((ext_vector_type(4)))  int    v4i;

#define NSP 2048
#define NCH 256

#ifndef __has_builtin
#define __has_builtin(x) 0
#endif
#if __has_builtin(__builtin_amdgcn_global_load_async_to_lds_b128)
#define HAVE_ASYNC_LDS 1
typedef __attribute__((address_space(1))) v4i* gv4i_p;   // global int4*
typedef __attribute__((address_space(3))) v4i* lv4i_p;   // LDS int4*
#else
#define HAVE_ASYNC_LDS 0
#endif

__device__ __forceinline__ bf16 f2bf(float f) {
  unsigned u = __builtin_bit_cast(unsigned, f);
  unsigned r = u + 0x7FFFu + ((u >> 16) & 1u);
  u16 h = (u16)(r >> 16);
  return __builtin_bit_cast(bf16, h);
}

// 16-byte global -> LDS copy; async DMA path when available.
__device__ __forceinline__ void g2l16(const void* g, void* l) {
#if HAVE_ASYNC_LDS
  __builtin_amdgcn_global_load_async_to_lds_b128(
      (gv4i_p)(unsigned long long)g,
      (lv4i_p)(unsigned long long)l, 0, 0);
#else
  *(bf16x8*)l = *(const bf16x8*)g;
#endif
}

__device__ __forceinline__ void wait_async_lds() {
#if HAVE_ASYNC_LDS
#if __has_builtin(__builtin_amdgcn_s_wait_asynccnt)
  __builtin_amdgcn_s_wait_asynccnt(0);
#endif
#endif
}

// A/B fragment for V_WMMA_F32_16X16X32_BF16; LDS rows are K-contiguous so
// each half is two ds_load_b128. kb = (lane>>4)*8 per ISA 16-bit layout.
__device__ __forceinline__ bf16x16 ldsfrag(const bf16* p, int kb) {
  bf16x8 lo = *(const bf16x8*)(p + kb);
  bf16x8 hi = *(const bf16x8*)(p + 16 + kb);
  bf16x16 r;
#pragma unroll
  for (int e = 0; e < 8; ++e) { r[e] = lo[e]; r[e + 8] = hi[e]; }
  return r;
}

__device__ __forceinline__ f32x8 wmma_bf16(bf16x16 a, bf16x16 b, f32x8 c) {
  return __builtin_amdgcn_wmma_f32_16x16x32_bf16(false, a, false, b,
                                                 (short)0, c, false, false);
}

// ---------------------------------------------------------------------------
// One-shot prep kernels (cheap, bandwidth-bound, run every launch)
// ---------------------------------------------------------------------------
__global__ __launch_bounds__(256)
void cvt_bf16(const float* __restrict__ x, bf16* __restrict__ y, int n) {
  for (int i = blockIdx.x * 256 + threadIdx.x; i < n; i += gridDim.x * 256)
    y[i] = f2bf(x[i]);
}

// conv weights -> bf16, layout [otile][tap][cchunk][row128][col32]
__global__ __launch_bounds__(256)
void pack_conv_w(const float* __restrict__ w, bf16* __restrict__ wp,
                 int Cin, int transposed) {
  const int kchunks = Cin / 32;
  const int total = 2 * 27 * kchunks * 128 * 32;
  for (int i = blockIdx.x * 256 + threadIdx.x; i < total; i += gridDim.x * 256) {
    int col = i & 31;
    int row = (i >> 5) & 127;
    int rem = i >> 12;
    int cc  = rem % kchunks;
    int rem2 = rem / kchunks;
    int tap = rem2 % 27;
    int ot  = rem2 / 27;
    int o = ot * 128 + row, c = cc * 32 + col;
    float val = transposed
        ? w[((size_t)c * NCH + o) * 27 + (26 - tap)]
        : w[((size_t)o * Cin + c) * 27 + tap];
    wp[i] = f2bf(val);
  }
}

// 256x256 GEMM weights -> bf16, layout [otile][cchunk][row128][col32]
__global__ __launch_bounds__(256)
void pack_gemm_w(const float* __restrict__ w, bf16* __restrict__ wp) {
  const int total = 2 * 8 * 128 * 32;
  for (int i = blockIdx.x * 256 + threadIdx.x; i < total; i += gridDim.x * 256) {
    int col = i & 31;
    int row = (i >> 5) & 127;
    int cc  = (i >> 12) & 7;
    int ot  = i >> 15;
    wp[i] = f2bf(w[(size_t)(ot * 128 + row) * NCH + cc * 32 + col]);
  }
}

// ---------------------------------------------------------------------------
// Implicit-GEMM conv3d k=3 SAME; packed bf16 weights, bf16 activations.
// Block tile 128 out-ch x 64 spatial; 8 waves x 4 WMMA per K-step.
// ---------------------------------------------------------------------------
__global__ __launch_bounds__(256)
void conv3d_wmma(const bf16* __restrict__ x, const bf16* __restrict__ wp,
                 const float* __restrict__ bias, const float* __restrict__ pos,
                 float* __restrict__ y, bf16* __restrict__ ybf, int Cin)
{
  __shared__ alignas(16) bf16 As[128 * 32];
  __shared__ alignas(16) bf16 Bs[64 * 32];

  const int t = threadIdx.x, lane = t & 31, wave = t >> 5;
  const int nbase = blockIdx.x * 64;
  const int otile = blockIdx.y;
  const int obase = otile * 128;
  const int b = blockIdx.z;
  const int kb = (lane >> 4) * 8, m15 = lane & 15;
  const u16* xs = (const u16*)x;

  f32x8 acc[4] = {};
  const int kchunks = Cin / 32;

  for (int tap = 0; tap < 27; ++tap) {
    const int dz = tap / 9 - 1;
    const int dy = (tap / 3) % 3 - 1;
    const int dx = tap % 3 - 1;
    for (int cc = 0; cc < kchunks; ++cc) {
      const int cbase = cc * 32;
      __syncthreads();
      // A tile: contiguous 8KB copy from packed weights
      const bf16* wt = wp + (size_t)((otile * 27 + tap) * kchunks + cc) * 4096;
      g2l16(wt + t * 8, As + t * 8);
      g2l16(wt + 2048 + t * 8, As + 2048 + t * 8);
      // B tile: zero-padded spatial gather, bf16 pairs along c
#pragma unroll
      for (int i = 0; i < 4; ++i) {
        int idx = t + i * 256;               // 1024 channel-pairs
        int nl = idx & 63, c = (idx >> 6) * 2;
        int n = nbase + nl;
        int d = (n >> 8) + dz, hh = ((n >> 4) & 15) + dy, ww = (n & 15) + dx;
        unsigned pk = 0;
        if (((unsigned)d < 8u) & ((unsigned)hh < 16u) & ((unsigned)ww < 16u)) {
          size_t base = ((size_t)b * Cin + cbase + c) * NSP +
                        ((d << 8) | (hh << 4) | ww);
          pk = (unsigned)xs[base] | ((unsigned)xs[base + NSP] << 16);
        }
        *(unsigned*)(void*)&Bs[nl * 32 + c] = pk;
      }
      wait_async_lds();
      __syncthreads();
      bf16x16 af = ldsfrag(&As[(wave * 16 + m15) * 32], kb);
#pragma unroll
      for (int j = 0; j < 4; ++j) {
        bf16x16 bfr = ldsfrag(&Bs[(j * 16 + m15) * 32], kb);
        acc[j] = wmma_bf16(af, bfr, acc[j]);
      }
    }
  }
  const int half8 = (lane >> 4) * 8;
#pragma unroll
  for (int j = 0; j < 4; ++j) {
    int n = nbase + j * 16 + m15;
#pragma unroll
    for (int r = 0; r < 8; ++r) {
      int o = obase + wave * 16 + half8 + r;
      float val = acc[j][r] + bias[o];
      if (pos) val += pos[(size_t)o * NSP + n];
      size_t off = ((size_t)b * NCH + o) * NSP + n;
      y[off] = val;
      if (ybf) ybf[off] = f2bf(val);
    }
  }
}

// ---------------------------------------------------------------------------
// Projection GEMM, packed bf16 weights + bf16 activations, bf16-only output.
// nmajor=1 writes [b,h,n,d] (for K so attention staging is contiguous).
// ---------------------------------------------------------------------------
__global__ __launch_bounds__(256)
void proj_gemm_wmma(const bf16* __restrict__ wp, const float* __restrict__ bias,
                    const bf16* __restrict__ x, bf16* __restrict__ y, int nmajor)
{
  __shared__ alignas(16) bf16 As[128 * 32];
  __shared__ alignas(16) bf16 Bs[64 * 32];
  const int t = threadIdx.x, lane = t & 31, wave = t >> 5;
  const int nbase = blockIdx.x * 64, otile = blockIdx.y, obase = otile * 128;
  const int b = blockIdx.z;
  const int kb = (lane >> 4) * 8, m15 = lane & 15;
  const u16* xs = (const u16*)x;

  f32x8 acc[4] = {};
  for (int cc = 0; cc < 8; ++cc) {
    const int cbase = cc * 32;
    __syncthreads();
    const bf16* wt = wp + (size_t)(otile * 8 + cc) * 4096;
    g2l16(wt + t * 8, As + t * 8);
    g2l16(wt + 2048 + t * 8, As + 2048 + t * 8);
#pragma unroll
    for (int i = 0; i < 4; ++i) {
      int idx = t + i * 256;
      int nl = idx & 63, c = (idx >> 6) * 2;
      size_t base = ((size_t)b * NCH + cbase + c) * NSP + nbase + nl;
      unsigned pk = (unsigned)xs[base] | ((unsigned)xs[base + NSP] << 16);
      *(unsigned*)(void*)&Bs[nl * 32 + c] = pk;
    }
    wait_async_lds();
    __syncthreads();
    bf16x16 af = ldsfrag(&As[(wave * 16 + m15) * 32], kb);
#pragma unroll
    for (int j = 0; j < 4; ++j) {
      bf16x16 bfr = ldsfrag(&Bs[(j * 16 + m15) * 32], kb);
      acc[j] = wmma_bf16(af, bfr, acc[j]);
    }
  }
  const int half8 = (lane >> 4) * 8;
#pragma unroll
  for (int j = 0; j < 4; ++j) {
    int n = nbase + j * 16 + m15;
#pragma unroll
    for (int r = 0; r < 8; ++r) {
      int o = obase + wave * 16 + half8 + r;
      float val = acc[j][r] + bias[o];
      if (nmajor) {
        int hh = o >> 5, dd = o & 31;
        y[((size_t)(b * 8 + hh) * NSP + n) * 32 + dd] = f2bf(val);
      } else {
        y[((size_t)b * NCH + o) * NSP + n] = f2bf(val);
      }
    }
  }
}

// ---------------------------------------------------------------------------
// Flash attention: q,v bf16 [b,h,d,n]; k bf16 [b,h,n,d] so both K/V LDS
// tiles stage as straight (async) 16B copies. Online softmax in f32.
// out = gamma * softmax(QK/sqrt(32)) V + xin  (f32 + bf16 copies).
// ---------------------------------------------------------------------------
__global__ __launch_bounds__(256)
void flash_attn_wmma(const bf16* __restrict__ q, const bf16* __restrict__ k,
                     const bf16* __restrict__ v, const float* __restrict__ xin,
                     const float* __restrict__ gamma,
                     float* __restrict__ out, bf16* __restrict__ outbf)
{
  __shared__ alignas(16) bf16 Ks[64 * 32];      // [key][d]
  __shared__ alignas(16) bf16 Vs[32 * 64];      // [d][key]
  __shared__ alignas(16) bf16 Ps[8][16 * 64];   // per-wave P [m][key]

  const int t = threadIdx.x, lane = t & 31, wave = t >> 5;
  const int m15 = lane & 15, kb = (lane >> 4) * 8;
  const int h = blockIdx.y, b = blockIdx.z;
  const int qbase = blockIdx.x * 128 + wave * 16;
  const size_t bh = (size_t)(b * 8 + h) * 32;     // channel base (d-major bufs)
  const bf16* kB = k + ((size_t)(b * 8 + h) * NSP) * 32;  // n-major buffer
  const float scale = 0.17677669529663689f;

  // persistent Q fragment (A layout: m=query, K=head-dim), direct bf16 reads
  bf16x16 qf;
#pragma unroll
  for (int e = 0; e < 16; ++e) {
    int d = (e < 8) ? (kb + e) : (16 + kb + (e - 8));
    qf[e] = q[(bh + d) * NSP + qbase + m15];
  }

  f32x8 o0 = {}, o1 = {};
  float rmax[8], rsum[8];
#pragma unroll
  for (int r = 0; r < 8; ++r) { rmax[r] = -1e30f; rsum[r] = 0.0f; }

  const int vd = t >> 3, vch = (t & 7) * 8;       // V staging coords

  for (int kt = 0; kt < 32; ++kt) {
    const int kb0 = kt * 64;
    __syncthreads();
    // K tile: 4KB contiguous; V tile: 32 rows x 128B
    g2l16(kB + (size_t)kb0 * 32 + t * 8, Ks + t * 8);
    g2l16(v + (bh + vd) * NSP + kb0 + vch, Vs + vd * 64 + vch);
    if (kt + 1 < 32) {
      __builtin_prefetch(kB + (size_t)(kb0 + 64) * 32 + t * 8, 0, 1);
      __builtin_prefetch(v + (bh + vd) * NSP + kb0 + 64 + vch, 0, 1);
    }
    wait_async_lds();
    __syncthreads();

    // S = (Q^T K) * scale
    f32x8 s[4];
#pragma unroll
    for (int j = 0; j < 4; ++j) {
      bf16x16 kf = ldsfrag(&Ks[(j * 16 + m15) * 32], kb);
      f32x8 z = {};
      s[j] = wmma_bf16(qf, kf, z);
#pragma unroll
      for (int r = 0; r < 8; ++r) s[j][r] *= scale;
    }

    // online softmax: row m = r + 8*(lane>>4), cols across 16 lanes
#pragma unroll
    for (int r = 0; r < 8; ++r) {
      float mloc = -1e30f;
#pragma unroll
      for (int j = 0; j < 4; ++j) mloc = fmaxf(mloc, s[j][r]);
#pragma unroll
      for (int off = 1; off < 16; off <<= 1)
        mloc = fmaxf(mloc, __shfl_xor(mloc, off, 32));
      float mnew = fmaxf(rmax[r], mloc);
      float corr = __expf(rmax[r] - mnew);
      rmax[r] = mnew;
      float psum = 0.0f;
#pragma unroll
      for (int j = 0; j < 4; ++j) {
        float p = __expf(s[j][r] - mnew);
        s[j][r] = p;
        psum += p;
      }
#pragma unroll
      for (int off = 1; off < 16; off <<= 1)
        psum += __shfl_xor(psum, off, 32);
      rsum[r] = rsum[r] * corr + psum;
      o0[r] *= corr;
      o1[r] *= corr;
      int m = r + ((lane >> 4) * 8);
#pragma unroll
      for (int j = 0; j < 4; ++j)
        Ps[wave][m * 64 + j * 16 + m15] = f2bf(s[j][r]);
    }

    // O += P V^T
#pragma unroll
    for (int kk = 0; kk < 2; ++kk) {
      bf16x16 pf  = ldsfrag(&Ps[wave][m15 * 64 + kk * 32], kb);
      bf16x16 vf0 = ldsfrag(&Vs[m15 * 64 + kk * 32], kb);
      bf16x16 vf1 = ldsfrag(&Vs[(16 + m15) * 64 + kk * 32], kb);
      o0 = wmma_bf16(pf, vf0, o0);
      o1 = wmma_bf16(pf, vf1, o1);
    }
  }

  const float g = gamma[0];
#pragma unroll
  for (int r = 0; r < 8; ++r) {
    int nq = qbase + r + ((lane >> 4) * 8);
    float inv = 1.0f / rsum[r];
#pragma unroll
    for (int dt = 0; dt < 2; ++dt) {
      float ov = (dt ? o1[r] : o0[r]) * inv;
      int ch = h * 32 + dt * 16 + m15;
      size_t off = ((size_t)b * NCH + ch) * NSP + nq;
      float val = g * ov + xin[off];
      out[off] = val;
      outbf[off] = f2bf(val);
    }
  }
}

// ---------------------------------------------------------------------------
extern "C" void kernel_launch(void* const* d_in, const int* in_sizes, int n_in,
                              void* d_out, int out_size, void* d_ws, size_t ws_size,
                              hipStream_t stream) {
  (void)in_sizes; (void)n_in; (void)out_size; (void)ws_size;
  const float* x1    = (const float*)d_in[1];
  const float* lat_w = (const float*)d_in[2];
  const float* lat_b = (const float*)d_in[3];
  const float* pos   = (const float*)d_in[4];
  const float* aw[6] = { (const float*)d_in[5],  (const float*)d_in[7],
                         (const float*)d_in[9],  (const float*)d_in[12],
                         (const float*)d_in[14], (const float*)d_in[16] };
  const float* ab[6] = { (const float*)d_in[6],  (const float*)d_in[8],
                         (const float*)d_in[10], (const float*)d_in[13],
                         (const float*)d_in[15], (const float*)d_in[17] };
  const float* a0_g  = (const float*)d_in[11];
  const float* a1_g  = (const float*)d_in[18];
  const float* out_w = (const float*)d_in[19];
  const float* out_b = (const float*)d_in[20];
  float* out = (float*)d_out;

  const size_t SZ = (size_t)2 * NCH * NSP;          // 1,048,576 activations
  char* p = (char*)d_ws;
  auto carve = [&](size_t bytes) {
    void* q = (void*)p; p += (bytes + 255) & ~(size_t)255; return q;
  };
  float* latA   = (float*)carve(SZ * 4);
  float* latB   = (float*)carve(SZ * 4);
  bf16*  latAbf = (bf16*)carve(SZ * 2);
  bf16*  latBbf = (bf16*)carve(SZ * 2);
  bf16*  xbf    = (bf16*)carve((size_t)2 * 512 * NSP * 2);
  bf16*  qbf    = (bf16*)carve(SZ * 2);
  bf16*  kbf    = (bf16*)carve(SZ * 2);
  bf16*  vbf    = (bf16*)carve(SZ * 2);
  bf16*  latWp  = (bf16*)carve((size_t)2 * 27 * 16 * 4096 * 2);
  bf16*  outWp  = (bf16*)carve((size_t)2 * 27 * 8 * 4096 * 2);
  bf16*  gwp[6];
  for (int i = 0; i < 6; ++i) gwp[i] = (bf16*)carve((size_t)2 * 8 * 4096 * 2);

  dim3 blk(256);
  dim3 gconv(32, 2, 2);
  dim3 gattn(16, 8, 2);

  // prep: bf16 activation copy + packed bf16 weights
  cvt_bf16<<<2048, blk, 0, stream>>>(x1, xbf, 2 * 512 * NSP);
  pack_conv_w<<<1024, blk, 0, stream>>>(lat_w, latWp, 512, 0);
  pack_conv_w<<<512, blk, 0, stream>>>(out_w, outWp, 256, 1);
  for (int i = 0; i < 6; ++i)
    pack_gemm_w<<<64, blk, 0, stream>>>(aw[i], gwp[i]);

  // lateral conv + pos
  conv3d_wmma<<<gconv, blk, 0, stream>>>(xbf, latWp, lat_b, pos,
                                         latA, latAbf, 512);
  // module 0
  proj_gemm_wmma<<<gconv, blk, 0, stream>>>(gwp[0], ab[0], latAbf, qbf, 0);
  proj_gemm_wmma<<<gconv, blk, 0, stream>>>(gwp[1], ab[1], latAbf, kbf, 1);
  proj_gemm_wmma<<<gconv, blk, 0, stream>>>(gwp[2], ab[2], latAbf, vbf, 0);
  flash_attn_wmma<<<gattn, blk, 0, stream>>>(qbf, kbf, vbf, latA, a0_g,
                                             latB, latBbf);
  // module 1
  proj_gemm_wmma<<<gconv, blk, 0, stream>>>(gwp[3], ab[3], latBbf, qbf, 0);
  proj_gemm_wmma<<<gconv, blk, 0, stream>>>(gwp[4], ab[4], latBbf, kbf, 1);
  proj_gemm_wmma<<<gconv, blk, 0, stream>>>(gwp[5], ab[5], latBbf, vbf, 0);
  flash_attn_wmma<<<gattn, blk, 0, stream>>>(qbf, kbf, vbf, latB, a1_g,
                                             latA, latAbf);
  // output transpose conv
  conv3d_wmma<<<gconv, blk, 0, stream>>>(latAbf, outWp, out_b, nullptr,
                                         out, nullptr, 256);
}
